// BLIP3oDiTBlock_19241453486844
// MI455X (gfx1250) — matrix operand order repaired and checked by reading, not compiled, not run on hardware
//
#include <hip/hip_runtime.h>

// ---------------------------------------------------------------------------
// DiT block: B=32, S=L=256, H=1024, NH=16, HD=64, I=4096, E=4096.
// All GEMMs via v_wmma_f32_16x16x32_bf16 (bf16 operands, f32 accumulate).
// GEMM: 128x128 block tile, BK=64, double-buffered LDS, 8 waves (wave32),
// each wave owns a 32x64 sub-tile (2x4 accumulators of v8f).
// ---------------------------------------------------------------------------

typedef __attribute__((ext_vector_type(2)))  float        f32x2;
typedef __attribute__((ext_vector_type(4)))  float        f32x4;
typedef __attribute__((ext_vector_type(4)))  unsigned int u32x4;
typedef __attribute__((ext_vector_type(8)))  float        v8f;
typedef __attribute__((ext_vector_type(16))) __bf16       v16bf;
typedef __attribute__((ext_vector_type(2)))  __bf16       v2bf;

union Frag {
    v16bf v;
    u32x4 q[2];
};

// f32 pair -> packed bf16 dword; __builtin_convertvector on a 2-vector
// pattern-matches to a single v_cvt_pk_bf16_f32.
__device__ __forceinline__ unsigned pk_bf16(float a, float b) {
    union { v2bf v; unsigned u; } x;
    f32x2 t;
    t.x = a; t.y = b;
    x.v = __builtin_convertvector(t, v2bf);
    return x.u;
}

// ---------------------------------------------------------------------------
// Generic tiled WMMA GEMM:  C[M,N] = epilogue(alpha * A[M,K] @ B)
//   TRANS_B == false : B is K x N row-major
//   TRANS_B == true  : B is N x K row-major (operand used as B^T)
// EPI: 0 = alpha*acc, 1 = acc + bias[n], 2 = resid + gate[m/div][n]*acc,
//      3 = resid + acc
// Batched over blockIdx.z: z -> (z/nh, z%nh) with separate pointer strides.
// Requires: K % 64 == 0 (holds for K = 64, 256, 1024, 4096); gateDiv % 8 == 0.
// ---------------------------------------------------------------------------
#define BM 128
#define BN 128
#define BK 64
#define LDSP 72   // padded bf16 row stride (64 data + 8 pad): conflict-free,
                  // 144 B = multiple of 16 so b128 LDS accesses stay aligned

template <int EPI, bool TRANS_B>
__global__ __launch_bounds__(256) void gemm_wmma(
    const float* __restrict__ A, const float* __restrict__ Bm,
    float* __restrict__ C,
    int M, int N, int K, int lda, int ldb, int ldc,
    long zAb, long zAh, long zBb, long zBh, long zCb, long zCh, int nh,
    float alpha,
    const float* __restrict__ bias,
    const float* __restrict__ resid,
    const float* __restrict__ gate, int gateLd, int gateDiv)
{
    __shared__ __align__(16) unsigned short As[2][BM * LDSP];
    __shared__ __align__(16) unsigned short Bs[2][BN * LDSP];

    const int z  = blockIdx.z;
    const int zb = z / nh;
    const int zh = z % nh;
    A  += (long)zb * zAb + (long)zh * zAh;
    Bm += (long)zb * zBb + (long)zh * zBh;
    C  += (long)zb * zCb + (long)zh * zCh;
    const float* residP = resid ? resid + (long)zb * zCb + (long)zh * zCh : resid;

    const int tid  = threadIdx.x;
    const int lane = tid & 31;
    const int wid  = tid >> 5;   // 8 waves
    const int wm   = wid & 3;    // 4 waves along M
    const int wn   = wid >> 2;   // 2 waves along N
    const int half = lane >> 4;
    const int r16  = lane & 15;

    const int blockM = blockIdx.y * BM;
    const int blockN = blockIdx.x * BN;

    v8f acc[2][4];
#pragma unroll
    for (int mi = 0; mi < 2; ++mi)
#pragma unroll
        for (int ni = 0; ni < 4; ++ni)
#pragma unroll
            for (int r = 0; r < 8; ++r) acc[mi][ni][r] = 0.0f;

    f32x4 pa[8], pb[8];

    auto loadA = [&](int k0) {
#pragma unroll
        for (int i = 0; i < 8; ++i) {
            int idx = tid + i * 256;          // 2048 float4 units: 128 x 16
            int row = idx >> 4;
            int c4  = (idx & 15) << 2;
            f32x4 v = {0.f, 0.f, 0.f, 0.f};
            int gr = blockM + row;
            if (gr < M) v = *(const f32x4*)(A + (long)gr * lda + k0 + c4);
            pa[i] = v;
        }
    };
    auto storeA = [&](unsigned short* dst) {
#pragma unroll
        for (int i = 0; i < 8; ++i) {
            int idx = tid + i * 256;
            int row = idx >> 4;
            int c4  = (idx & 15) << 2;
            unsigned* p = (unsigned*)(dst + row * LDSP + c4);
            p[0] = pk_bf16(pa[i].x, pa[i].y);
            p[1] = pk_bf16(pa[i].z, pa[i].w);
        }
    };
    // Non-transposed B mapping: col-group = tid&31, base = tid>>5;
    // iteration i covers krow = base*2 + (i&1) + (i>>1)*16, so that i pairs
    // (2p, 2p+1) give ADJACENT k rows of the same column -> packed b32 stores.
    auto loadB = [&](int k0) {
        if (TRANS_B) {
#pragma unroll
            for (int i = 0; i < 8; ++i) {
                int idx  = tid + i * 256;
                int nrow = idx >> 4;
                int c4   = (idx & 15) << 2;
                f32x4 v = {0.f, 0.f, 0.f, 0.f};
                int gn = blockN + nrow;
                if (gn < N) v = *(const f32x4*)(Bm + (long)gn * ldb + k0 + c4);
                pb[i] = v;
            }
        } else {
            int col  = tid & 31;
            int base = tid >> 5;
            int gc   = blockN + (col << 2);
#pragma unroll
            for (int i = 0; i < 8; ++i) {
                int krow = base * 2 + (i & 1) + (i >> 1) * 16;
                f32x4 v = {0.f, 0.f, 0.f, 0.f};
                if (gc < N) v = *(const f32x4*)(Bm + (long)(k0 + krow) * ldb + gc);
                pb[i] = v;
            }
        }
    };
    auto storeB = [&](unsigned short* dst) {
        if (TRANS_B) {
#pragma unroll
            for (int i = 0; i < 8; ++i) {
                int idx  = tid + i * 256;
                int nrow = idx >> 4;
                int c4   = (idx & 15) << 2;
                unsigned* p = (unsigned*)(dst + nrow * LDSP + c4);
                p[0] = pk_bf16(pb[i].x, pb[i].y);
                p[1] = pk_bf16(pb[i].z, pb[i].w);
            }
        } else {
            int col  = tid & 31;
            int base = tid >> 5;
            int nloc = col << 2;
#pragma unroll
            for (int p = 0; p < 4; ++p) {
                int krow0 = base * 2 + p * 16;       // even -> 4B aligned
                f32x4 v0 = pb[2 * p];
                f32x4 v1 = pb[2 * p + 1];
                *(unsigned*)(dst + (nloc + 0) * LDSP + krow0) = pk_bf16(v0.x, v1.x);
                *(unsigned*)(dst + (nloc + 1) * LDSP + krow0) = pk_bf16(v0.y, v1.y);
                *(unsigned*)(dst + (nloc + 2) * LDSP + krow0) = pk_bf16(v0.z, v1.z);
                *(unsigned*)(dst + (nloc + 3) * LDSP + krow0) = pk_bf16(v0.w, v1.w);
            }
        }
    };

    const int nk = K / BK;
    loadA(0);
    loadB(0);

    for (int kt = 0; kt < nk; ++kt) {
        const int cur = kt & 1;
        storeA(&As[cur][0]);
        storeB(&Bs[cur][0]);
        __syncthreads();

        if (kt + 1 < nk) {
            loadA((kt + 1) * BK);
            loadB((kt + 1) * BK);
        }
        if (kt + 2 < nk) {   // speculative L2 prefetch of the k+2 tiles
            __builtin_prefetch(A + (long)(blockM + (tid >> 1)) * lda + (kt + 2) * BK, 0, 0);
            if (TRANS_B)
                __builtin_prefetch(Bm + (long)(blockN + (tid >> 1)) * ldb + (kt + 2) * BK, 0, 0);
            else
                __builtin_prefetch(Bm + (long)((kt + 2) * BK + (tid >> 3)) * ldb + blockN, 0, 0);
        }

        // ---- 2 K-steps of WMMA from LDS tile `cur` ----
#pragma unroll
        for (int ks = 0; ks < 2; ++ks) {
            const int ko = ks * 32;
            // A 16x32 bf16 fragment: lane half h: VGPR0-3 = K[8h..8h+7],
            //                                    VGPR4-7 = K[16+8h..23+8h]
            Frag af[2];
#pragma unroll
            for (int mi = 0; mi < 2; ++mi) {
                int row = wm * 32 + mi * 16 + r16;
                af[mi].q[0] = *(const u32x4*)(&As[cur][row * LDSP + ko + 8 * half]);
                af[mi].q[1] = *(const u32x4*)(&As[cur][row * LDSP + ko + 16 + 8 * half]);
            }
            // B 32x16 bf16 fragment: lane half h: VGPR0-3 = K[8h..8h+7], VGPR4-7 = K[16+8h..23+8h]
            Frag bfr[4];
#pragma unroll
            for (int ni = 0; ni < 4; ++ni) {
                int nrow = wn * 64 + ni * 16 + r16;
                bfr[ni].q[0] = *(const u32x4*)(&Bs[cur][nrow * LDSP + ko + 8 * half]);
                bfr[ni].q[1] = *(const u32x4*)(&Bs[cur][nrow * LDSP + ko + 16 + 8 * half]);
            }
#pragma unroll
            for (int mi = 0; mi < 2; ++mi)
#pragma unroll
                for (int ni = 0; ni < 4; ++ni)
                    acc[mi][ni] = __builtin_amdgcn_wmma_f32_16x16x32_bf16(
                        false, af[mi].v, false, bfr[ni].v, (short)0,
                        acc[mi][ni], false, false);
        }
        __syncthreads();
    }

    // ---- epilogue: C/D layout: VGPR r -> M = r + 8*half, N = lane%16 ----
    // An 8-row span starting at rowBase (multiple of 8) never crosses a
    // gateDiv boundary (gateDiv % 8 == 0), so gate/bias values are hoisted.
#pragma unroll
    for (int mi = 0; mi < 2; ++mi) {
        const int  rowBase = blockM + wm * 32 + mi * 16 + 8 * half;
        const bool fullM   = (rowBase + 7) < M;
#pragma unroll
        for (int ni = 0; ni < 4; ++ni) {
            const int gCol = blockN + wn * 64 + ni * 16 + r16;
            if (gCol >= N) continue;
            float bv = 0.0f, gv = 0.0f;
            if (EPI == 1) bv = bias[gCol];
            if (EPI == 2) gv = gate[(long)(rowBase / gateDiv) * gateLd + gCol];
            float* cp = C + (long)rowBase * ldc + gCol;
            const float* rp = (EPI >= 2) ? residP + (long)rowBase * ldc + gCol
                                         : nullptr;
            if (fullM) {
#pragma unroll
                for (int r = 0; r < 8; ++r) {
                    float v = acc[mi][ni][r] * alpha;
                    if (EPI == 1) v += bv;
                    if (EPI == 2) v = *rp + gv * v;
                    if (EPI == 3) v = *rp + v;
                    *cp = v;
                    cp += ldc;
                    if (EPI >= 2) rp += ldc;
                }
            } else {
#pragma unroll
                for (int r = 0; r < 8; ++r) {
                    if (rowBase + r < M) {
                        float v = acc[mi][ni][r] * alpha;
                        if (EPI == 1) v += bv;
                        if (EPI == 2) v = rp[(long)r * ldc] + gv * v;
                        if (EPI == 3) v = rp[(long)r * ldc] + v;
                        cp[(long)r * ldc] = v;
                    }
                }
            }
        }
    }
}

// ---------------------------------------------------------------------------
// Elementwise / reduction kernels
// ---------------------------------------------------------------------------
__device__ __forceinline__ float blockReduceSum(float v, float* red) {
    int t = threadIdx.x;
    red[t] = v; __syncthreads();
    for (int s = 128; s > 0; s >>= 1) {
        if (t < s) red[t] += red[t + s];
        __syncthreads();
    }
    float r = red[0]; __syncthreads();
    return r;
}
__device__ __forceinline__ float blockReduceMax(float v, float* red) {
    int t = threadIdx.x;
    red[t] = v; __syncthreads();
    for (int s = 128; s > 0; s >>= 1) {
        if (t < s) red[t] = fmaxf(red[t], red[t + s]);
        __syncthreads();
    }
    float r = red[0]; __syncthreads();
    return r;
}

__global__ __launch_bounds__(256) void silu_k(const float* __restrict__ x,
                                              float* __restrict__ y, int n) {
    int i = blockIdx.x * 256 + threadIdx.x;
    if (i < n) {
        float v = x[i];
        y[i] = v / (1.0f + __expf(-v));
    }
}

__global__ __launch_bounds__(256) void silumul_k(const float* __restrict__ g,
                                                 const float* __restrict__ u,
                                                 float* __restrict__ o, long n) {
    long i = (long)blockIdx.x * 256 + threadIdx.x;
    if (i < n) {
        float v = g[i];
        o[i] = (v / (1.0f + __expf(-v))) * u[i];
    }
}

// Row kernel over (B*S) rows of H=1024:
//   y = rms(x, w) [ * (1+sc[b]) + sh[b] ] [ then 2D RoPE (GRID=16, qd=128) ]
template <bool HAS_MOD, bool HAS_ROPE>
__global__ __launch_bounds__(256) void rmsmod_k(
    const float* __restrict__ x, const float* __restrict__ w,
    const float* __restrict__ sc, const float* __restrict__ sh, int modLd,
    float* __restrict__ out, int Srows)
{
    __shared__ float buf[1024];
    __shared__ float red[256];
    const long row = blockIdx.x;
    const int  b   = (int)(row / Srows);
    const int  s   = (int)(row % Srows);
    const float* xr = x + row * 1024;
    const int t = threadIdx.x;

    float vals[4];
    float ss = 0.0f;
#pragma unroll
    for (int j = 0; j < 4; ++j) {
        int c = t + j * 256;
        float v = xr[c];
        vals[j] = v;
        ss += v * v;
    }
    ss = blockReduceSum(ss, red);
    float scale = rsqrtf(ss * (1.0f / 1024.0f) + 1e-6f);

#pragma unroll
    for (int j = 0; j < 4; ++j) {
        int c = t + j * 256;
        float v = vals[j] * scale * w[c];
        if (HAS_MOD)
            v = v * (1.0f + sc[(long)b * modLd + c]) + sh[(long)b * modLd + c];
        if (HAS_ROPE) buf[c] = v;
        else out[row * 1024 + c] = v;
    }
    if (HAS_ROPE) {
        __syncthreads();
        float px = (float)(s >> 4);   // pos_x = s // 16
        float py = (float)(s & 15);   // pos_y = s % 16
#pragma unroll
        for (int j = 0; j < 4; ++j) {
            int c = t + j * 256;
            float o;
            if (c < 512) {
                int seg = c >> 7;
                int i   = c & 127;
                // inv[i] = 10000^(-i/128)
                float inv = __expf(-(float)i * (9.210340371976184f / 128.0f));
                if (seg == 0)
                    o = buf[i] * __cosf(px * inv) - buf[i + 128] * __sinf(px * inv);
                else if (seg == 1)
                    o = buf[i] * __sinf(px * inv) + buf[i + 128] * __cosf(px * inv);
                else if (seg == 2)
                    o = buf[256 + i] * __cosf(py * inv) - buf[384 + i] * __sinf(py * inv);
                else
                    o = buf[256 + i] * __sinf(py * inv) + buf[384 + i] * __cosf(py * inv);
            } else {
                o = buf[c];
            }
            out[row * 1024 + c] = o;
        }
    }
}

// Row softmax over length-256 rows (one block per row, one element/thread).
__global__ __launch_bounds__(256) void softmax_k(float* __restrict__ p) {
    __shared__ float red[256];
    const long row = blockIdx.x;
    float* r = p + row * 256;
    int t = threadIdx.x;
    float v = r[t];
    float m = blockReduceMax(v, red);
    float e = __expf(v - m);
    float s = blockReduceSum(e, red);
    r[t] = e / s;
}

// ---------------------------------------------------------------------------
// Host-side launch helper
// ---------------------------------------------------------------------------
template <int EPI, bool TB>
static void gemmL(hipStream_t st, const float* A, const float* B, float* C,
                  int M, int N, int K, int lda, int ldb, int ldc,
                  long zAb = 0, long zAh = 0, long zBb = 0, long zBh = 0,
                  long zCb = 0, long zCh = 0, int nh = 1, int Z = 1,
                  float alpha = 1.0f, const float* bias = nullptr,
                  const float* resid = nullptr, const float* gate = nullptr,
                  int gateLd = 0, int gateDiv = 8)
{
    dim3 grid((N + BN - 1) / BN, (M + BM - 1) / BM, Z);
    gemm_wmma<EPI, TB><<<grid, 256, 0, st>>>(A, B, C, M, N, K, lda, ldb, ldc,
                                             zAb, zAh, zBb, zBh, zCb, zCh, nh,
                                             alpha, bias, resid, gate, gateLd,
                                             gateDiv);
}

extern "C" void kernel_launch(void* const* d_in, const int* in_sizes, int n_in,
                              void* d_out, int out_size, void* d_ws, size_t ws_size,
                              hipStream_t stream) {
    (void)in_sizes; (void)n_in; (void)out_size; (void)ws_size;
    const int  Bb = 32, S = 256, Lc = 256, H = 1024, NH = 16, HD = 64, I = 4096, E = 4096;
    const int  BS = Bb * S;   // 8192
    const int  BL = Bb * Lc;  // 8192

    const float* hidden = (const float*)d_in[0];
    const float* enc    = (const float*)d_in[1];
    const float* temb   = (const float*)d_in[2];
    const float* wq1    = (const float*)d_in[3];
    const float* wk1    = (const float*)d_in[4];
    const float* wv1    = (const float*)d_in[5];
    const float* wo1    = (const float*)d_in[6];
    const float* wq2    = (const float*)d_in[7];
    const float* wk2    = (const float*)d_in[8];
    const float* wv2    = (const float*)d_in[9];
    const float* wo2    = (const float*)d_in[10];
    const float* eva_w  = (const float*)d_in[11];
    const float* eva_b  = (const float*)d_in[12];
    const float* ada_w  = (const float*)d_in[13];
    const float* ada_b  = (const float*)d_in[14];
    const float* gate_w = (const float*)d_in[15];
    const float* up_w   = (const float*)d_in[16];
    const float* down_w = (const float*)d_in[17];
    const float* n1     = (const float*)d_in[18];
    const float* n2     = (const float*)d_in[19];
    const float* n3     = (const float*)d_in[20];
    float* out = (float*)d_out;

    // workspace layout (float offsets, all 16B aligned)
    float* ws  = (float*)d_ws;
    float* st  = ws;                    // 32*1024
    float* mod = ws + 32768;            // 32*6144
    float* eva = ws + 229376;           // 8192*1024
    float* xb  = ws + 8617984;          // 8192*1024 (reused: x_mod, h2, y)
    float* q   = ws + 17006592;         // 8192*1024
    float* kk  = ws + 25395200;         // 8192*1024
    float* vv  = ws + 33783808;         // 8192*1024
    float* ao  = ws + 42172416;         // 8192*1024
    float* h   = ws + 50561024;         // 8192*1024
    float* sc  = ws + 58949632;         // 32*16*256*256 (reused as MLP gate g)
    float* uu  = ws + 92504064;         // 8192*4096

    const float iscale = 0.125f;        // HD^-0.5 = 1/8
    const long  SL = (long)S * Lc;

    // 1) st = silu(timestep_emb)
    silu_k<<<(Bb * H + 255) / 256, 256, 0, stream>>>(temb, st, Bb * H);
    // 2) mod = st @ ada_w + ada_b                      (32 x 6144, K=1024)
    gemmL<1, false>(stream, st, ada_w, mod, Bb, 6 * H, H, H, 6 * H, 6 * H,
                    0, 0, 0, 0, 0, 0, 1, 1, 1.0f, ada_b);
    // 3) eva = enc @ eva_w + eva_b                     (8192 x 1024, K=4096)
    gemmL<1, false>(stream, enc, eva_w, eva, BL, H, E, E, H, H,
                    0, 0, 0, 0, 0, 0, 1, 1, 1.0f, eva_b);
    // 4) xb = rope(rms(hidden,n1)*(1+sc_msa)+sh_msa)
    rmsmod_k<true, true><<<BS, 256, 0, stream>>>(hidden, n1, mod + H, mod, 6 * H, xb, S);
    // 5) q/k/v projections (self-attn)
    gemmL<0, false>(stream, xb, wq1, q,  BS, H, H, H, H, H);
    gemmL<0, false>(stream, xb, wk1, kk, BS, H, H, H, H, H);
    gemmL<0, false>(stream, xb, wv1, vv, BS, H, H, H, H, H);
    // 6) scores = (q . k^T) / 8   batched over (b,h)
    gemmL<0, true>(stream, q, kk, sc, S, Lc, HD, H, H, Lc,
                   (long)S * H, HD, (long)Lc * H, HD, (long)NH * SL, SL,
                   NH, Bb * NH, iscale);
    // 7) softmax rows
    softmax_k<<<Bb * NH * S, 256, 0, stream>>>(sc);
    // 8) ao = probs @ v
    gemmL<0, false>(stream, sc, vv, ao, S, HD, Lc, Lc, H, H,
                    (long)NH * SL, SL, (long)Lc * H, HD, (long)S * H, HD,
                    NH, Bb * NH);
    // 9) h = hidden + g_msa * (ao @ wo1)
    gemmL<2, false>(stream, ao, wo1, h, BS, H, H, H, H, H,
                    0, 0, 0, 0, 0, 0, 1, 1, 1.0f, nullptr, hidden,
                    mod + 2 * H, 6 * H, S);
    // 10) xb = rms(h, n2)
    rmsmod_k<false, false><<<BS, 256, 0, stream>>>(h, n2, nullptr, nullptr, 0, xb, S);
    // 11) cross-attn projections
    gemmL<0, false>(stream, xb,  wq2, q,  BS, H, H, H, H, H);
    gemmL<0, false>(stream, eva, wk2, kk, BL, H, H, H, H, H);
    gemmL<0, false>(stream, eva, wv2, vv, BL, H, H, H, H, H);
    // 12) scores
    gemmL<0, true>(stream, q, kk, sc, S, Lc, HD, H, H, Lc,
                   (long)S * H, HD, (long)Lc * H, HD, (long)NH * SL, SL,
                   NH, Bb * NH, iscale);
    // 13) softmax
    softmax_k<<<Bb * NH * S, 256, 0, stream>>>(sc);
    // 14) ao = probs @ v
    gemmL<0, false>(stream, sc, vv, ao, S, HD, Lc, Lc, H, H,
                    (long)NH * SL, SL, (long)Lc * H, HD, (long)S * H, HD,
                    NH, Bb * NH);
    // 15) h = h + ao @ wo2   (in-place residual: each element read then written
    //     by the same thread only)
    gemmL<3, false>(stream, ao, wo2, h, BS, H, H, H, H, H,
                    0, 0, 0, 0, 0, 0, 1, 1, 1.0f, nullptr, h);
    // 16) xb = rms(h, n3)*(1+sc_mlp)+sh_mlp
    rmsmod_k<true, false><<<BS, 256, 0, stream>>>(h, n3, mod + 4 * H, mod + 3 * H,
                                                  6 * H, xb, S);
    // 17) g = xb @ gate_w ; u = xb @ up_w   (8192 x 4096, K=1024)
    gemmL<0, false>(stream, xb, gate_w, sc, BS, I, H, H, I, I);
    gemmL<0, false>(stream, xb, up_w,   uu, BS, I, H, H, I, I);
    // 18) g = silu(g) * u
    {
        long n = (long)BS * I;
        silumul_k<<<(unsigned)((n + 255) / 256), 256, 0, stream>>>(sc, uu, sc, n);
    }
    // 19) out = h + g_mlp * (g @ down_w)
    gemmL<2, false>(stream, sc, down_w, out, BS, H, I, I, H, H,
                    0, 0, 0, 0, 0, 0, 1, 1, 1.0f, nullptr, h,
                    mod + 5 * H, 6 * H, S);
}
